// SkillPathGNN_6640019440479
// MI455X (gfx1250) — matrix operand order repaired and checked by reading, not compiled, not run on hardware
//
#include <hip/hip_runtime.h>

typedef __attribute__((ext_vector_type(16))) _Float16 v16h;
typedef __attribute__((ext_vector_type(8)))  float    v8f;

// ---------------------------------------------------------------------------
// float atomic max via signed/unsigned int ordering trick (correct for all
// IEEE orderings incl. mixed signs)
// ---------------------------------------------------------------------------
__device__ __forceinline__ void atomicMaxF(float* addr, float v) {
    if (v >= 0.0f) atomicMax((int*)addr, __float_as_int(v));
    else           atomicMin((unsigned int*)addr, __float_as_uint(v));
}

// ---------------------------------------------------------------------------
// WMMA GEMM: C[M x NN] = A[M x KK] @ B[KK x NN] (+bias, +relu)
// A,B,C row-major f32; compute in f16 WMMA with f32 accumulate.
// B is staged transposed into LDS as f16 (<=32KB; WGP has 320KB).
// One 16x16 output tile per wave, 8 waves per 256-thread block.
// ---------------------------------------------------------------------------
template <int KK, int NN, bool BIAS, bool RELU>
__global__ void __launch_bounds__(256) gemm_wmma(
    const float* __restrict__ A, const float* __restrict__ B,
    const float* __restrict__ bias, float* __restrict__ C, int M)
{
    __shared__ __align__(32) _Float16 sB[KK * NN];   // sB[c*KK + k] = B[k][c]
    for (int idx = threadIdx.x; idx < KK * NN; idx += 256) {
        int k = idx / NN, c = idx % NN;
        sB[c * KK + k] = (_Float16)B[idx];
    }
    __syncthreads();

    const int lane = threadIdx.x & 31;
    const int wv   = threadIdx.x >> 5;
    const int numColTiles = NN / 16;
    const int totalTiles  = (M / 16) * numColTiles;
    const int tile = blockIdx.x * 8 + wv;            // wave-uniform guard
    if (tile >= totalTiles) return;
    const int rowBase = (tile / numColTiles) * 16;
    const int colBase = (tile % numColTiles) * 16;
    const int l  = lane & 15;
    const int hi = lane >> 4;

    v8f acc = {};
    const float*    arow = A  + (size_t)(rowBase + l) * KK;
    const _Float16* bcol = sB + (colBase + l) * KK;

    for (int kb = 0; kb < KK; kb += 32) {
        if (kb + 32 < KK) __builtin_prefetch(arow + kb + 32, 0, 1);
        // A 16x32 f16 fragment: lane(l,hi) holds row M=l,
        // K = hi*16 + {0..15}  (one contiguous run)
        const float* ap = arow + kb + hi * 16;
        float4 q0 = *(const float4*)(ap);
        float4 q1 = *(const float4*)(ap + 4);
        float4 q2 = *(const float4*)(ap + 8);
        float4 q3 = *(const float4*)(ap + 12);
        v16h a;
        a[0]=(_Float16)q0.x;  a[1]=(_Float16)q0.y;  a[2]=(_Float16)q0.z;  a[3]=(_Float16)q0.w;
        a[4]=(_Float16)q1.x;  a[5]=(_Float16)q1.y;  a[6]=(_Float16)q1.z;  a[7]=(_Float16)q1.w;
        a[8]=(_Float16)q2.x;  a[9]=(_Float16)q2.y;  a[10]=(_Float16)q2.z; a[11]=(_Float16)q2.w;
        a[12]=(_Float16)q3.x; a[13]=(_Float16)q3.y; a[14]=(_Float16)q3.z; a[15]=(_Float16)q3.w;
        // B 32x16 fragment: lane l = column, K = hi*16 + j  (contiguous in sB)
        v16h b = *(const v16h*)(bcol + kb + hi * 16);
        acc = __builtin_amdgcn_wmma_f32_16x16x32_f16(
                false, a, false, b, (short)0, acc, false, false);
    }

    // C/D layout: VGPR r -> row = rowBase + hi*8 + r, col = colBase + l
    #pragma unroll
    for (int r = 0; r < 8; ++r) {
        int row = rowBase + hi * 8 + r;
        int col = colBase + l;
        float v = acc[r];
        if (BIAS) v += bias[col];
        if (RELU) v = fmaxf(v, 0.0f);
        C[(size_t)row * NN + col] = v;
    }
}

// ---------------------------------------------------------------------------
// Degree / normalization
// ---------------------------------------------------------------------------
__global__ void deg_init_k(float* deg, int N) {
    int i = blockIdx.x * blockDim.x + threadIdx.x;
    if (i < N) deg[i] = 1.0f;                         // self loop
}
__global__ void deg_accum_k(const int* __restrict__ dstA, float* deg, int E) {
    int e = blockIdx.x * blockDim.x + threadIdx.x;
    if (e < E) atomicAdd(&deg[dstA[e]], 1.0f);
}
__global__ void dinv_k(float* deg, int N) {
    int i = blockIdx.x * blockDim.x + threadIdx.x;
    if (i < N) { float d = deg[i]; deg[i] = (d > 0.f) ? rsqrtf(d) : 0.f; }
}

// ---------------------------------------------------------------------------
// GCN aggregation: agg = D^-1/2 (A+I) D^-1/2 * hpre ; then +bias, relu
// ---------------------------------------------------------------------------
__global__ void gcn_self_k(const float* __restrict__ hpre,
                           const float* __restrict__ dinv,
                           float* __restrict__ agg, int N) {
    int t = blockIdx.x * blockDim.x + threadIdx.x;
    if (t >= N * 64) return;
    int n = t >> 6;
    float di = dinv[n];
    agg[t] = hpre[t] * di * di;
}
__global__ void gcn_edge_k(const float* __restrict__ hpre,
                           const float* __restrict__ dinv,
                           const int* __restrict__ srcA,
                           const int* __restrict__ dstA,
                           float* __restrict__ agg, int E) {
    int t = blockIdx.x * blockDim.x + threadIdx.x;
    if (t >= E * 64) return;
    int e = t >> 6, c = t & 63;
    int s = srcA[e], d = dstA[e];
    float nrm = dinv[s] * dinv[d];
    atomicAdd(&agg[d * 64 + c], hpre[s * 64 + c] * nrm);
}
__global__ void gcn_post_k(float* __restrict__ agg,
                           const float* __restrict__ b, int N) {
    int t = blockIdx.x * blockDim.x + threadIdx.x;
    if (t >= N * 64) return;
    agg[t] = fmaxf(agg[t] + b[t & 63], 0.0f);
}

// ---------------------------------------------------------------------------
// GAT pieces
// ---------------------------------------------------------------------------
__global__ void att_logits_k(const float* __restrict__ hp,
                             const float* __restrict__ att_s,
                             const float* __restrict__ att_d,
                             float* asrc, float* adst,
                             float* amax, float* denm, int N) {
    int t = blockIdx.x * blockDim.x + threadIdx.x;
    if (t >= N * 4) return;
    int n = t >> 2, h = t & 3;
    const float* v  = hp + (size_t)n * 256 + h * 64;
    const float* ws = att_s + h * 64;
    const float* wd = att_d + h * 64;
    float s0 = 0.f, s1 = 0.f;
    #pragma unroll 8
    for (int d = 0; d < 64; ++d) { s0 += v[d] * ws[d]; s1 += v[d] * wd[d]; }
    asrc[t] = s0; adst[t] = s1;
    amax[t] = -1e30f; denm[t] = 0.f;
}
__global__ void zero_k(float* p, int n) {
    int t = blockIdx.x * blockDim.x + threadIdx.x;
    if (t < n) p[t] = 0.f;
}
__device__ __forceinline__ void edge_sd(int e, int E,
                                        const int* srcA, const int* dstA,
                                        int& s, int& d) {
    if (e < E) { s = srcA[e]; d = dstA[e]; }
    else       { s = e - E;   d = e - E;   }          // self loop
}
__global__ void gat_max_k(const float* __restrict__ asrc,
                          const float* __restrict__ adst,
                          const int* __restrict__ srcA,
                          const int* __restrict__ dstA,
                          float* amax, int E, int N) {
    int t = blockIdx.x * blockDim.x + threadIdx.x;
    int E2 = E + N;
    if (t >= E2 * 4) return;
    int e = t >> 2, h = t & 3, s, d;
    edge_sd(e, E, srcA, dstA, s, d);
    float a = asrc[s * 4 + h] + adst[d * 4 + h];
    a = (a > 0.f) ? a : 0.2f * a;                     // leaky relu 0.2
    atomicMaxF(&amax[d * 4 + h], a);
}
__global__ void gat_expsum_k(const float* __restrict__ asrc,
                             const float* __restrict__ adst,
                             const int* __restrict__ srcA,
                             const int* __restrict__ dstA,
                             const float* __restrict__ amax,
                             float* denm, float* ew, int E, int N) {
    int t = blockIdx.x * blockDim.x + threadIdx.x;
    int E2 = E + N;
    if (t >= E2 * 4) return;
    int e = t >> 2, h = t & 3, s, d;
    edge_sd(e, E, srcA, dstA, s, d);
    float a = asrc[s * 4 + h] + adst[d * 4 + h];
    a = (a > 0.f) ? a : 0.2f * a;
    float ex = __expf(a - amax[d * 4 + h]);
    ew[t] = ex;
    atomicAdd(&denm[d * 4 + h], ex);
}
__global__ void gat_scatter_k(const float* __restrict__ hp,
                              const float* __restrict__ ew,
                              const float* __restrict__ denm,
                              const int* __restrict__ srcA,
                              const int* __restrict__ dstA,
                              float* __restrict__ outh, int E, int N) {
    int t = blockIdx.x * blockDim.x + threadIdx.x;   // (E+N)*256 threads
    int E2 = E + N;
    if (t >= E2 * 256) return;
    int e = t >> 8, r = t & 255, h = r >> 6, s, d;
    edge_sd(e, E, srcA, dstA, s, d);
    float coef = ew[e * 4 + h] / denm[d * 4 + h];
    atomicAdd(&outh[(size_t)d * 256 + r], hp[(size_t)s * 256 + r] * coef);
}
__global__ void gat_bias_k(float* __restrict__ outh,
                           const float* __restrict__ bg, int N) {
    int t = blockIdx.x * blockDim.x + threadIdx.x;
    if (t >= N * 256) return;
    outh[t] += bg[t & 255];
}

// ---------------------------------------------------------------------------
// Final: out[n] = h[n,:] . Wf2 + bf2
// ---------------------------------------------------------------------------
__global__ void final_k(const float* __restrict__ h,
                        const float* __restrict__ Wf2,
                        const float* __restrict__ bf2,
                        float* __restrict__ out, int N) {
    int n = blockIdx.x * blockDim.x + threadIdx.x;
    if (n >= N) return;
    float acc = bf2[0];
    #pragma unroll 8
    for (int d = 0; d < 64; ++d) acc += h[(size_t)n * 64 + d] * Wf2[d];
    out[n] = acc;
}

// ---------------------------------------------------------------------------
static inline unsigned cdivu(long long a, long long b) { return (unsigned)((a + b - 1) / b); }

extern "C" void kernel_launch(void* const* d_in, const int* in_sizes, int n_in,
                              void* d_out, int out_size, void* d_ws, size_t ws_size,
                              hipStream_t stream) {
    const float* x    = (const float*)d_in[0];
    const int*   eidx = (const int*)d_in[1];          // [2,E] int32 (JAX x64 off)
    const float* W1 = (const float*)d_in[2];
    const float* b1 = (const float*)d_in[3];
    const float* W2 = (const float*)d_in[4];
    const float* b2 = (const float*)d_in[5];
    const float* Wg = (const float*)d_in[6];
    const float* att_src = (const float*)d_in[7];
    const float* att_dst = (const float*)d_in[8];
    const float* bg  = (const float*)d_in[9];
    const float* Wf1 = (const float*)d_in[10];
    const float* bf1 = (const float*)d_in[11];
    const float* Wf2 = (const float*)d_in[12];
    const float* bf2 = (const float*)d_in[13];
    float* out = (float*)d_out;

    const int N  = in_sizes[0] / 256;
    const int E  = in_sizes[1] / 2;
    const int E2 = E + N;
    const int* srcA = eidx;
    const int* dstA = eidx + E;

    // workspace layout (floats): ~145 MB total
    float* w = (float*)d_ws;
    float* dinv = w;  w += N;
    float* bufA = w;  w += (size_t)N * 64;            // GEMM pre-agg / MLP hidden
    float* bufB = w;  w += (size_t)N * 64;            // aggregated h
    float* hp   = w;  w += (size_t)N * 256;           // GAT projected features
    float* asrc = w;  w += (size_t)N * 4;
    float* adst = w;  w += (size_t)N * 4;
    float* amax = w;  w += (size_t)N * 4;
    float* denm = w;  w += (size_t)N * 4;
    float* ew   = w;  w += (size_t)E2 * 4;            // per-edge exp weights
    float* bufE = w;  w += (size_t)N * 256;           // GAT output

    const unsigned T = 256;

    // degrees + D^-1/2
    deg_init_k <<<cdivu(N, T), T, 0, stream>>>(dinv, N);
    deg_accum_k<<<cdivu(E, T), T, 0, stream>>>(dstA, dinv, E);
    dinv_k     <<<cdivu(N, T), T, 0, stream>>>(dinv, N);

    // ---- GCN layer 1 ----
    gemm_wmma<256, 64, false, false><<<cdivu((long long)(N/16) * 4, 8), T, 0, stream>>>(x, W1, nullptr, bufA, N);
    gcn_self_k<<<cdivu((long long)N * 64, T), T, 0, stream>>>(bufA, dinv, bufB, N);
    gcn_edge_k<<<cdivu((long long)E * 64, T), T, 0, stream>>>(bufA, dinv, srcA, dstA, bufB, E);
    gcn_post_k<<<cdivu((long long)N * 64, T), T, 0, stream>>>(bufB, b1, N);

    // ---- GCN layer 2 ----
    gemm_wmma<64, 64, false, false><<<cdivu((long long)(N/16) * 4, 8), T, 0, stream>>>(bufB, W2, nullptr, bufA, N);
    gcn_self_k<<<cdivu((long long)N * 64, T), T, 0, stream>>>(bufA, dinv, bufB, N);
    gcn_edge_k<<<cdivu((long long)E * 64, T), T, 0, stream>>>(bufA, dinv, srcA, dstA, bufB, E);
    gcn_post_k<<<cdivu((long long)N * 64, T), T, 0, stream>>>(bufB, b2, N);

    // ---- GAT ----
    gemm_wmma<64, 256, false, false><<<cdivu((long long)(N/16) * 16, 8), T, 0, stream>>>(bufB, Wg, nullptr, hp, N);
    att_logits_k<<<cdivu((long long)N * 4, T), T, 0, stream>>>(hp, att_src, att_dst, asrc, adst, amax, denm, N);
    zero_k      <<<cdivu((long long)N * 256, T), T, 0, stream>>>(bufE, N * 256);
    gat_max_k   <<<cdivu((long long)E2 * 4, T), T, 0, stream>>>(asrc, adst, srcA, dstA, amax, E, N);
    gat_expsum_k<<<cdivu((long long)E2 * 4, T), T, 0, stream>>>(asrc, adst, srcA, dstA, amax, denm, ew, E, N);
    gat_scatter_k<<<cdivu((long long)E2 * 256, T), T, 0, stream>>>(hp, ew, denm, srcA, dstA, bufE, E, N);
    gat_bias_k  <<<cdivu((long long)N * 256, T), T, 0, stream>>>(bufE, bg, N);

    // ---- MLP head ----
    gemm_wmma<256, 64, true, true><<<cdivu((long long)(N/16) * 4, 8), T, 0, stream>>>(bufE, Wf1, bf1, bufA, N);
    final_k<<<cdivu(N, T), T, 0, stream>>>(bufA, Wf2, bf2, out, N);
}
